// CondEquiUpdate_17179869184407
// MI455X (gfx1250) — matrix operand order repaired and checked by reading, not compiled, not run on hardware
//
#include <hip/hip_runtime.h>
#include <hip/hip_bf16.h>

#define N_NODES   50000
#define N_EDGES_C 800000
#define HID       128
#define EDGE_D    64
#define DIST_D    32
#define TIME_D    128
#define IN_CH     352           // 2*HID + EDGE_D + DIST_D
#define ET        64            // edges per block
#define SA_STRIDE 360           // padded bf16 row stride (720B = 180 words, conflict-free)
#define SX_STRIDE 136           // padded bf16 row stride (272B = 68 words, conflict-free)
#define LN_EPS    1e-6f
#define COORS_EPS 1e-8f

typedef __attribute__((ext_vector_type(16))) __bf16 v16bf;
typedef __attribute__((ext_vector_type(8)))  float  v8f;

union BFrag {
    v16bf v;
    uint4 q[2];
};

union PkBF2 {
    uint u;
    __bf16 h[2];
};

__device__ __forceinline__ uint pk_bf2(float x, float y) {
    PkBF2 p;
    p.h[0] = (__bf16)x;
    p.h[1] = (__bf16)y;
    return p.u;
}

__device__ __forceinline__ float siluf(float x) {
    return x / (1.0f + __expf(-x));
}

// ---------------------------------------------------------------------------
// Prologue: f32 -> bf16 weight conversion (W_in: 128x352, W1: 128x128)
// ---------------------------------------------------------------------------
__global__ void convert_weights_kernel(const float* __restrict__ Win,
                                       const float* __restrict__ W1,
                                       __bf16* __restrict__ Win_bf,
                                       __bf16* __restrict__ W1_bf) {
    const int total = HID * IN_CH + HID * HID;
    for (int i = blockIdx.x * blockDim.x + threadIdx.x; i < total;
         i += gridDim.x * blockDim.x) {
        if (i < HID * IN_CH) {
            Win_bf[i] = (__bf16)Win[i];
        } else {
            int j = i - HID * IN_CH;
            W1_bf[j] = (__bf16)W1[j];
        }
    }
}

// ---------------------------------------------------------------------------
// Prologue: t = silu(time_emb) @ W_time^T + b_time  -> shift/scale [256]
// ---------------------------------------------------------------------------
__global__ void time_cond_kernel(const float* __restrict__ te,
                                 const float* __restrict__ Wt,
                                 const float* __restrict__ bt,
                                 float* __restrict__ ss) {
    __shared__ float st[TIME_D];
    const int t = threadIdx.x;  // 256 threads
    if (t < TIME_D) st[t] = siluf(te[t]);
    __syncthreads();
    float s = 0.0f;
    const float* w = Wt + t * TIME_D;
#pragma unroll 8
    for (int k = 0; k < TIME_D; ++k) s += st[k] * w[k];
    ss[t] = s + bt[t];
}

// ---------------------------------------------------------------------------
// Prologue: out = pos
// ---------------------------------------------------------------------------
__global__ void init_out_kernel(const float* __restrict__ pos,
                                float* __restrict__ out) {
    int i = blockIdx.x * blockDim.x + threadIdx.x;
    if (i < N_NODES * 3) out[i] = pos[i];
}

// ---------------------------------------------------------------------------
// Main edge kernel: 64 edges/block, 4 waves, one 16-row tile per wave.
// All 4 waves stream the same W_in k-slices in lockstep -> WGP$ provides the
// weight reuse; registers stay under 256 so the scheduler can pipeline B
// loads deep ahead of the WMMAs (no vgpr-msb churn, no accumulator copies).
// ---------------------------------------------------------------------------
__global__ __launch_bounds__(128)
void edge_update_kernel(const float* __restrict__ h,
                        const float* __restrict__ pos,
                        const int*   __restrict__ ei,
                        const float* __restrict__ edge_attr,
                        const float* __restrict__ dist,
                        const float* __restrict__ coors_scale,
                        const float* __restrict__ b_in,
                        const float* __restrict__ b1,
                        const float* __restrict__ W2,
                        const float* __restrict__ shiftscale,  // ws: [256]
                        const __bf16* __restrict__ Win_bf,     // ws: [128][352]
                        const __bf16* __restrict__ W1_bf,      // ws: [128][128]
                        float* __restrict__ out) {
    __shared__ __bf16 sA[ET * SA_STRIDE];   // 46080 B
    __shared__ __bf16 sX[ET * SX_STRIDE];   // 17408 B

    const int tid  = threadIdx.x;
    const int e0   = blockIdx.x * ET;
    const int lane = tid & 31;
    const int wave = tid >> 5;     // 0..3: 16-row tile
    const int lhi  = lane >> 4;    // 0/1
    const int llo  = lane & 15;

    // ---- Stage A tile: [ET][IN_CH] = concat(h[row], h[col], edge_attr, dist)
    // 2 threads per edge row; packed bf16x4 (8B) LDS stores.
    {
        const int er   = tid >> 1;   // 0..63 edge within block
        const int half = tid & 1;    // covers 176 of the 352 elements
        const int e    = e0 + er;
        __bf16* dst = &sA[er * SA_STRIDE];
        if (e < N_EDGES_C) {
            const int r = ei[e];
            const int c = ei[N_EDGES_C + e];
            const float* hr = h + (size_t)r * HID;
            const float* hc = h + (size_t)c * HID;
            const float* ea = edge_attr + (size_t)e * EDGE_D;
            const float* dd = dist + (size_t)e * DIST_D;
#pragma unroll
            for (int k = half * 64; k < half * 64 + 64; k += 4) {
                float4 v = *(const float4*)(hr + k);
                uint2 w = {pk_bf2(v.x, v.y), pk_bf2(v.z, v.w)};
                *(uint2*)(dst + k) = w;
            }
#pragma unroll
            for (int k = half * 64; k < half * 64 + 64; k += 4) {
                float4 v = *(const float4*)(hc + k);
                uint2 w = {pk_bf2(v.x, v.y), pk_bf2(v.z, v.w)};
                *(uint2*)(dst + HID + k) = w;
            }
#pragma unroll
            for (int k = half * 32; k < half * 32 + 32; k += 4) {
                float4 v = *(const float4*)(ea + k);
                uint2 w = {pk_bf2(v.x, v.y), pk_bf2(v.z, v.w)};
                *(uint2*)(dst + 2 * HID + k) = w;
            }
#pragma unroll
            for (int k = half * 16; k < half * 16 + 16; k += 4) {
                float4 v = *(const float4*)(dd + k);
                uint2 w = {pk_bf2(v.x, v.y), pk_bf2(v.z, v.w)};
                *(uint2*)(dst + 2 * HID + EDGE_D + k) = w;
            }
        } else {
            uint2 z = {0u, 0u};
            for (int k = half * 176; k < half * 176 + 176; k += 4)
                *(uint2*)(dst + k) = z;
        }
    }
    __syncthreads();

    // ---- GEMM1: x[16x128] = A[16x352] @ W_in^T, f32 accumulate
    v8f acc[8];
    {
        v8f z = {0.f, 0.f, 0.f, 0.f, 0.f, 0.f, 0.f, 0.f};
#pragma unroll
        for (int nt = 0; nt < 8; ++nt) acc[nt] = z;
    }

    const __bf16* aBase = &sA[(wave * 16 + llo) * SA_STRIDE];
    for (int k0 = 0; k0 < IN_CH; k0 += 32) {
        BFrag a;  // lanes 0-15: K=k0..k0+7, k0+16..k0+23 ; lanes 16-31: +8
        a.q[0] = *(const uint4*)(aBase + k0 + lhi * 8);
        a.q[1] = *(const uint4*)(aBase + k0 + lhi * 8 + 16);
#pragma unroll
        for (int nt = 0; nt < 8; ++nt) {
            const __bf16* bp = Win_bf + (size_t)(nt * 16 + llo) * IN_CH + k0 + lhi * 8;
            BFrag b;  // lane = column, K same as A: k0+8h..+7, k0+16+8h..+7
            b.q[0] = *(const uint4*)(bp);
            b.q[1] = *(const uint4*)(bp + 16);
            acc[nt] = __builtin_amdgcn_wmma_f32_16x16x32_bf16(
                false, a.v, false, b.v, (short)0, acc[nt], false, false);
        }
    }

    // ---- + b_in, LayerNorm (row across 16 lanes), FiLM modulate, store bf16
#pragma unroll
    for (int nt = 0; nt < 8; ++nt) {
        const float bv = b_in[nt * 16 + llo];
#pragma unroll
        for (int i = 0; i < 8; ++i) acc[nt][i] += bv;
    }

    float mean_[8], rstd_[8];
#pragma unroll
    for (int i = 0; i < 8; ++i) {
        float s = 0.0f;
#pragma unroll
        for (int nt = 0; nt < 8; ++nt) s += acc[nt][i];
        for (int m = 1; m < 16; m <<= 1) s += __shfl_xor(s, m, 32);
        const float mu = s * (1.0f / HID);
        float v = 0.0f;
#pragma unroll
        for (int nt = 0; nt < 8; ++nt) {
            float d = acc[nt][i] - mu;
            v += d * d;
        }
        for (int m = 1; m < 16; m <<= 1) v += __shfl_xor(v, m, 32);
        mean_[i] = mu;
        rstd_[i] = rsqrtf(v * (1.0f / HID) + LN_EPS);
    }

#pragma unroll
    for (int nt = 0; nt < 8; ++nt) {
        const int c = nt * 16 + llo;
        const float sc = 1.0f + shiftscale[HID + c];
        const float sh = shiftscale[c];
#pragma unroll
        for (int i = 0; i < 8; ++i) {
            const int m = i + 8 * lhi;  // row within tile
            float xm = (acc[nt][i] - mean_[i]) * rstd_[i] * sc + sh;
            sX[(wave * 16 + m) * SX_STRIDE + c] = (__bf16)xm;
        }
    }
    __syncthreads();

    // ---- GEMM2: z[16x128] = x @ W1^T, f32 accumulate
    {
        v8f z = {0.f, 0.f, 0.f, 0.f, 0.f, 0.f, 0.f, 0.f};
#pragma unroll
        for (int nt = 0; nt < 8; ++nt) acc[nt] = z;
    }
    const __bf16* xBase = &sX[(wave * 16 + llo) * SX_STRIDE];
    for (int k0 = 0; k0 < HID; k0 += 32) {
        BFrag a;
        a.q[0] = *(const uint4*)(xBase + k0 + lhi * 8);
        a.q[1] = *(const uint4*)(xBase + k0 + lhi * 8 + 16);
#pragma unroll
        for (int nt = 0; nt < 8; ++nt) {
            const __bf16* bp = W1_bf + (size_t)(nt * 16 + llo) * HID + k0 + lhi * 8;
            BFrag b;
            b.q[0] = *(const uint4*)(bp);
            b.q[1] = *(const uint4*)(bp + 16);
            acc[nt] = __builtin_amdgcn_wmma_f32_16x16x32_bf16(
                false, a.v, false, b.v, (short)0, acc[nt], false, false);
        }
    }

    // ---- silu(z + b1) . W2  (per-row reduction across 16 lanes) -> tanh gate
    float p[8];
#pragma unroll
    for (int i = 0; i < 8; ++i) p[i] = 0.0f;
#pragma unroll
    for (int nt = 0; nt < 8; ++nt) {
        const float b1v = b1[nt * 16 + llo];
        const float w2v = W2[nt * 16 + llo];
#pragma unroll
        for (int i = 0; i < 8; ++i) {
            float z = acc[nt][i] + b1v;
            p[i] += siluf(z) * w2v;
        }
    }
#pragma unroll
    for (int i = 0; i < 8; ++i) {
        for (int m = 1; m < 16; m <<= 1) p[i] += __shfl_xor(p[i], m, 32);
    }

    // ---- coordinate update: lanes with (lane&8)==0 each own one edge-row
    if ((lane & 8) == 0) {
        const int i = lane & 7;
        const int m = i + 8 * lhi;
        const int e = e0 + wave * 16 + m;
        if (e < N_EDGES_C) {
            float pv = 0.0f;
#pragma unroll
            for (int j = 0; j < 8; ++j)
                if (j == i) pv = p[j];
            const float gate = tanhf(pv);
            const int r = ei[e];
            const int c = ei[N_EDGES_C + e];
            const float dx = pos[3 * r + 0] - pos[3 * c + 0];
            const float dy = pos[3 * r + 1] - pos[3 * c + 1];
            const float dz = pos[3 * r + 2] - pos[3 * c + 2];
            const float nrm = sqrtf(dx * dx + dy * dy + dz * dz);
            const float s = coors_scale[0] / fmaxf(nrm, COORS_EPS) * gate;
            atomicAdd(&out[3 * r + 0], dx * s);
            atomicAdd(&out[3 * r + 1], dy * s);
            atomicAdd(&out[3 * r + 2], dz * s);
        }
    }
}

// ---------------------------------------------------------------------------
extern "C" void kernel_launch(void* const* d_in, const int* in_sizes, int n_in,
                              void* d_out, int out_size, void* d_ws, size_t ws_size,
                              hipStream_t stream) {
    const float* h           = (const float*)d_in[0];
    const float* pos         = (const float*)d_in[1];
    const int*   edge_index  = (const int*)  d_in[2];
    const float* edge_attr   = (const float*)d_in[3];
    const float* dist        = (const float*)d_in[4];
    const float* time_emb    = (const float*)d_in[5];
    const float* coors_scale = (const float*)d_in[6];
    const float* W_time      = (const float*)d_in[7];
    const float* b_time      = (const float*)d_in[8];
    const float* W_in        = (const float*)d_in[9];
    const float* b_in        = (const float*)d_in[10];
    const float* W1          = (const float*)d_in[11];
    const float* b1          = (const float*)d_in[12];
    const float* W2          = (const float*)d_in[13];
    float* out = (float*)d_out;

    // workspace layout
    float*  ss     = (float*)d_ws;                                    // 256 f32
    __bf16* Win_bf = (__bf16*)((char*)d_ws + 1024);                   // 90112 B
    __bf16* W1_bf  = (__bf16*)((char*)d_ws + 1024 + HID * IN_CH * 2); // 32768 B

    convert_weights_kernel<<<240, 256, 0, stream>>>(W_in, W1, Win_bf, W1_bf);
    time_cond_kernel<<<1, 256, 0, stream>>>(time_emb, W_time, b_time, ss);
    init_out_kernel<<<(N_NODES * 3 + 255) / 256, 256, 0, stream>>>(pos, out);

    const int nblocks = (N_EDGES_C + ET - 1) / ET;  // 12500
    edge_update_kernel<<<nblocks, 128, 0, stream>>>(
        h, pos, edge_index, edge_attr, dist, coors_scale,
        b_in, b1, W2, ss, Win_bf, W1_bf, out);
}
